// TemporalLSTM_33517924778009
// MI455X (gfx1250) — matrix operand order repaired and checked by reading, not compiled, not run on hardware
//
#include <hip/hip_runtime.h>

#define Dd 512
#define NBATCH 256
#define TSTEPS 512
#define H4 2048

typedef __attribute__((ext_vector_type(16))) __bf16 v16bf;
typedef __attribute__((ext_vector_type(8)))  float  v8f;

static __device__ inline __bf16 f2bf(float f) {
  unsigned u = __builtin_bit_cast(unsigned, f);
  unsigned r = u + 0x7FFFu + ((u >> 16) & 1u);   // round-to-nearest-even
  unsigned short h = (unsigned short)(r >> 16);
  return __builtin_bit_cast(__bf16, h);
}

static __device__ inline float fsigmoid(float x) { return 1.f / (1.f + __expf(-x)); }
static __device__ inline float ftanh(float x) {
  x = fminf(15.f, fmaxf(-15.f, x));
  float e = __expf(2.f * x);
  return (e - 1.f) / (e + 1.f);
}

// ---------------------------------------------------------------------------
// Zero a float buffer
// ---------------------------------------------------------------------------
__global__ void zero_f32(float* __restrict__ p, int n) {
  int i = blockIdx.x * blockDim.x + threadIdx.x;
  if (i < n) p[i] = 0.f;
}

// ---------------------------------------------------------------------------
// Pack a [2048 x 512] fp32 weight matrix into bf16 WMMA B-fragment order:
// P[(((nt*16 + kb)*32) + half*16 + nl)*16 + e]  where n=nt*16+nl, k=kb*32+half*16+e
// (contiguous k order used here: lanes 0-15 hold K=0..15, lanes 16-31 hold K=16..31,
//  two K values per VGPR ascending)
// ---------------------------------------------------------------------------
__global__ void pack_weights(const float* __restrict__ W, unsigned short* __restrict__ Pu) {
  __bf16* P = (__bf16*)Pu;
  int idx = blockIdx.x * blockDim.x + threadIdx.x;   // over H4*Dd
  if (idx >= H4 * Dd) return;
  int n = idx >> 9;
  int k = idx & 511;
  int nt = n >> 4, nl = n & 15;
  int kb = k >> 5, half = (k >> 4) & 1, e = k & 15;
  P[(((nt * 16 + kb) * 32) + half * 16 + nl) * 16 + e] = f2bf(W[idx]);
}

__global__ void add_bias(const float* __restrict__ a, const float* __restrict__ b,
                         float* __restrict__ o) {
  int i = blockIdx.x * blockDim.x + threadIdx.x;
  if (i < H4) o[i] = a[i] + b[i];
}

// ---------------------------------------------------------------------------
// One LSTM timestep for one layer.
// grid = (8 d-blocks, 8 m-blocks), block = 256 threads (8 waves).
// Block computes gates for 32 rows x (4 gates x 64 d-cols), then fused
// elementwise update. Layer-1 also accumulates the masked pooling sum.
// K-loop is software-pipelined: fragments for iteration k+1 are loaded
// before the WMMAs of iteration k are issued.
// ---------------------------------------------------------------------------
__global__ __launch_bounds__(256) void lstm_step(
    const float* __restrict__ inp, int inp_stride,   // x_t (stride T*D) or h0 (stride D)
    const float* __restrict__ h_prev,
    float* __restrict__ c_state,
    float* __restrict__ h_out,
    const unsigned short* __restrict__ WihPu,
    const unsigned short* __restrict__ WhhPu,
    const float* __restrict__ bias,
    float* __restrict__ pool,                        // nullptr for layer 0
    const int* __restrict__ lengths,
    int t)
{
  const __bf16* WihP = (const __bf16*)WihPu;
  const __bf16* WhhP = (const __bf16*)WhhPu;

  // 64KB LDS: two 32KB A-fragment-packed activation tiles; reused as gate
  // staging (32 x 260 fp32, padded) after the K-loop.
  __shared__ __align__(16) __bf16 smem[2 * 32 * 512];
  __bf16* Ax = smem;
  __bf16* Ah = smem + 32 * 512;
  float*  gates = (float*)smem;

  const int tid  = threadIdx.x;
  const int lane = tid & 31;
  const int w    = tid >> 5;
  const int d0   = blockIdx.x * 64;   // d-range within each gate
  const int m0   = blockIdx.y * 32;   // batch-row base

  // ---- Stage x_t and h_prev rows into LDS, fp32 -> bf16, in WMMA A-fragment
  //      order (contiguous k order used here: lanes 0-15 K 0..15, lanes 16-31 K 16..31)
  for (int s = 0; s < 2; ++s) {
    const float* src = s ? h_prev : inp;
    const long   stride = s ? (long)Dd : (long)inp_stride;
    __bf16* dst = s ? Ah : Ax;
    for (int it = 0; it < 16; ++it) {
      int q = (it * 256 + tid) * 4;    // element index over 32*512
      int m = q >> 9;
      int k = q & 511;
      const float4 v = *(const float4*)(src + (long)(m0 + m) * stride + k);
      float vv[4] = {v.x, v.y, v.z, v.w};
      int ml = m & 15, mt = m >> 4;
#pragma unroll
      for (int j2 = 0; j2 < 4; ++j2) {
        int kk   = k + j2;
        int kb   = kk >> 5;
        int koff = kk & 31;
        int lane2 = ml + ((koff >> 4) << 4);
        int e     = koff & 15;
        dst[(((mt * 16 + kb) * 32) + lane2) * 16 + e] = f2bf(vv[j2]);
      }
    }
  }
  __syncthreads();

  // ---- WMMA GEMM: gates = x_t @ Wih^T + h @ Whh^T  (bf16 in, f32 accum)
  const int mt = w & 1;        // which 16-row half
  const int wn = w >> 1;       // 0..3, 64-col strip

  const __bf16* Abase[2] = { Ax + (mt * 16) * 32 * 16,
                             Ah + (mt * 16) * 32 * 16 };
  // Per-j weight base pointers for both phases (ih then hh)
  const __bf16* Wj[2][4];
#pragma unroll
  for (int j = 0; j < 4; ++j) {
    int ntj = wn * 4 + j;                          // local ntile 0..15
    int gi  = ntj >> 2;                            // gate index 0..3
    int ntg = gi * 32 + (d0 >> 4) + (ntj & 3);     // global 16-col tile in [0,128)
    Wj[0][j] = WihP + ntg * (16 * 32 * 16);
    Wj[1][j] = WhhP + ntg * (16 * 32 * 16);
  }

  v8f acc[4] = {};

  // Prologue: fragments for iteration 0
  v16bf a_cur = *(const v16bf*)(Abase[0] + lane * 16);
  v16bf b_cur[4];
#pragma unroll
  for (int j = 0; j < 4; ++j)
    b_cur[j] = *(const v16bf*)(Wj[0][j] + lane * 16);

#pragma unroll
  for (int it = 0; it < 32; ++it) {
    v16bf a_nxt = a_cur;
    v16bf b_nxt[4] = { b_cur[0], b_cur[1], b_cur[2], b_cur[3] };
    const int nxt = it + 1;
    if (nxt < 32) {                    // resolved at compile time (full unroll)
      const int ph = nxt >> 4;
      const int kb = nxt & 15;
      a_nxt = *(const v16bf*)(Abase[ph] + (kb * 32 + lane) * 16);
#pragma unroll
      for (int j = 0; j < 4; ++j)
        b_nxt[j] = *(const v16bf*)(Wj[ph][j] + (kb * 32 + lane) * 16);
    }
#pragma unroll
    for (int j = 0; j < 4; ++j) {
      acc[j] = __builtin_amdgcn_wmma_f32_16x16x32_bf16(
          false, a_cur, false, b_cur[j], (short)0, acc[j], false, false);
    }
    a_cur = a_nxt;
#pragma unroll
    for (int j = 0; j < 4; ++j) b_cur[j] = b_nxt[j];
  }

  __syncthreads();   // A tiles dead; reuse LDS for gate staging

  // C/D 16x16 f32 layout: VGPR r, lanes 0-15 -> M=r ; lanes 16-31 -> M=8+r
  {
    int row_half = (lane >> 4) * 8;
    int cl = lane & 15;
#pragma unroll
    for (int j = 0; j < 4; ++j) {
      int col = (wn * 4 + j) * 16 + cl;
#pragma unroll
      for (int r = 0; r < 8; ++r) {
        int row = mt * 16 + row_half + r;
        gates[row * 260 + col] = acc[j][r];
      }
    }
  }
  __syncthreads();

  // ---- Fused elementwise: gate nonlinearities, cell/hidden update, pooling
  for (int it = 0; it < 8; ++it) {
    int q   = it * 256 + tid;     // 0..2047 over 32 rows x 64 d
    int row = q >> 6;
    int dd  = q & 63;
    int gcol = d0 + dd;
    float gi_ = gates[row * 260 +   0 + dd] + bias[       gcol];
    float gf_ = gates[row * 260 +  64 + dd] + bias[ 512 + gcol];
    float gg_ = gates[row * 260 + 128 + dd] + bias[1024 + gcol];
    float go_ = gates[row * 260 + 192 + dd] + bias[1536 + gcol];
    float i = fsigmoid(gi_);
    float f = fsigmoid(gf_);
    float g = ftanh(gg_);
    float o = fsigmoid(go_);
    long gidx = (long)(m0 + row) * Dd + gcol;
    float c = f * c_state[gidx] + i * g;
    c_state[gidx] = c;
    float h = o * ftanh(c);
    h_out[gidx] = h;
    if (pool != nullptr && t < lengths[m0 + row]) pool[gidx] += h;
  }
}

// ---------------------------------------------------------------------------
// out = (pooled / len) @ Wout^T + bout    (67 MFLOP, negligible)
// ---------------------------------------------------------------------------
__global__ __launch_bounds__(256) void final_proj(
    const float* __restrict__ pooled, const int* __restrict__ lengths,
    const float* __restrict__ Wout, const float* __restrict__ bout,
    float* __restrict__ out)
{
  __shared__ float prow[Dd];
  int row = blockIdx.x;
  float inv = 1.f / (float)lengths[row];
  for (int k = threadIdx.x; k < Dd; k += 256) prow[k] = pooled[(long)row * Dd + k] * inv;
  __syncthreads();
  for (int j = threadIdx.x; j < Dd; j += 256) {
    const float* wr = Wout + (long)j * Dd;
    float s = 0.f;
#pragma unroll 4
    for (int k = 0; k < Dd; k += 4) {
      s += prow[k] * wr[k] + prow[k + 1] * wr[k + 1] +
           prow[k + 2] * wr[k + 2] + prow[k + 3] * wr[k + 3];
    }
    out[(long)row * Dd + j] = s + bout[j];
  }
}

// ---------------------------------------------------------------------------
extern "C" void kernel_launch(void* const* d_in, const int* in_sizes, int n_in,
                              void* d_out, int out_size, void* d_ws, size_t ws_size,
                              hipStream_t stream) {
  const float* x    = (const float*)d_in[0];
  const int*   lens = (const int*)d_in[1];
  const float* Wih0 = (const float*)d_in[2];
  const float* Whh0 = (const float*)d_in[3];
  const float* bih0 = (const float*)d_in[4];
  const float* bhh0 = (const float*)d_in[5];
  const float* Wih1 = (const float*)d_in[6];
  const float* Whh1 = (const float*)d_in[7];
  const float* bih1 = (const float*)d_in[8];
  const float* bhh1 = (const float*)d_in[9];
  const float* Wout = (const float*)d_in[10];
  const float* bout = (const float*)d_in[11];
  float* out = (float*)d_out;

  const int ND = NBATCH * Dd;          // 131072
  char* p = (char*)d_ws;
  float* h0a    = (float*)p; p += (size_t)ND * 4;
  float* h0b    = (float*)p; p += (size_t)ND * 4;
  float* c0     = (float*)p; p += (size_t)ND * 4;
  float* h1a    = (float*)p; p += (size_t)ND * 4;
  float* h1b    = (float*)p; p += (size_t)ND * 4;
  float* c1     = (float*)p; p += (size_t)ND * 4;
  float* pooled = (float*)p; p += (size_t)ND * 4;
  unsigned short* P_ih0 = (unsigned short*)p; p += (size_t)H4 * Dd * 2;
  unsigned short* P_hh0 = (unsigned short*)p; p += (size_t)H4 * Dd * 2;
  unsigned short* P_ih1 = (unsigned short*)p; p += (size_t)H4 * Dd * 2;
  unsigned short* P_hh1 = (unsigned short*)p; p += (size_t)H4 * Dd * 2;
  float* b0 = (float*)p; p += (size_t)H4 * 4;
  float* b1 = (float*)p; p += (size_t)H4 * 4;

  // States + pooled are contiguous: one zeroing pass (re-done every call,
  // deterministic under graph replay).
  zero_f32<<<(7 * ND + 255) / 256, 256, 0, stream>>>(h0a, 7 * ND);

  const int WN = H4 * Dd;
  pack_weights<<<(WN + 255) / 256, 256, 0, stream>>>(Wih0, P_ih0);
  pack_weights<<<(WN + 255) / 256, 256, 0, stream>>>(Whh0, P_hh0);
  pack_weights<<<(WN + 255) / 256, 256, 0, stream>>>(Wih1, P_ih1);
  pack_weights<<<(WN + 255) / 256, 256, 0, stream>>>(Whh1, P_hh1);
  add_bias<<<H4 / 256, 256, 0, stream>>>(bih0, bhh0, b0);
  add_bias<<<H4 / 256, 256, 0, stream>>>(bih1, bhh1, b1);

  dim3 grid(8, 8), blk(256);
  for (int t = 0; t < TSTEPS; ++t) {
    const float* h0p = (t & 1) ? h0b : h0a;
    float*       h0n = (t & 1) ? h0a : h0b;
    const float* h1p = (t & 1) ? h1b : h1a;
    float*       h1n = (t & 1) ? h1a : h1b;
    // layer 0: input = x[:, t, :] (row stride T*D)
    lstm_step<<<grid, blk, 0, stream>>>(x + (long)t * Dd, TSTEPS * Dd,
                                        h0p, c0, h0n, P_ih0, P_hh0, b0,
                                        nullptr, lens, t);
    // layer 1: input = h0[t] (row stride D); fused masked pooling
    lstm_step<<<grid, blk, 0, stream>>>(h0n, Dd,
                                        h1p, c1, h1n, P_ih1, P_hh1, b1,
                                        pooled, lens, t);
  }

  final_proj<<<NBATCH, 256, 0, stream>>>(pooled, lens, Wout, bout, out);
}
